// cheb_conv_with_SAt_7584912245217
// MI455X (gfx1250) — matrix operand order repaired and checked by reading, not compiled, not run on hardware
//
#include <hip/hip_runtime.h>
#include <hip/hip_bf16.h>

// Problem constants (from reference): B=8, N=1024, F=64, T=12, K=3, O=64
#define CB   8
#define CN   1024
#define CF   64
#define CT   12
#define CK   3
#define CO   64
#define CFT  (CF * CT)     // 768
#define CNT  (CN * CT)     // 12288 rows (n,t) per batch
#define CKF  (CK * CF)     // 192 reduction dim for stage 2

typedef __attribute__((ext_vector_type(16))) _Float16 v16h;
typedef __attribute__((ext_vector_type(8)))  _Float16 v8h;
typedef __attribute__((ext_vector_type(8)))  float    v8f;

// Workspace rhs layout: [b][(n*T+t)][(k*F+f)]  f16   (8 x 12288 x 192)
// -> stage-2 A fragments become two contiguous 16B loads per lane.

// ---------------------------------------------------------------------------
// Stage 1: rhs[b,(n,t),(k,f)] = sum_m (cheb[k,m,n]*att[b,m,n]) * x[b,m,f*T+t]
// Per (k,b): A = W^T (1024x1024), B = X (1024x768), f32 WMMA accumulate.
// WG tile 128n x 128c, 8 waves (4x2), wave tile 32x64 -> 2x4 accumulators,
// 8 WMMAs/wave/chunk.
// Pipeline: next chunk is fetched as RAW f32 into registers (pure load block,
// no consuming VALU -> loadcnt waits sink past the WMMA burst); the cheb*att
// multiply and f16 convert happen only at LDS-commit time of the next
// iteration. LDS commits are 16B ds_store_b128.
// ---------------------------------------------------------------------------
__global__ __launch_bounds__(256)
void cheb_stage1_wmma(const float* __restrict__ gs,    // (B,N,F,T)
                      const float* __restrict__ att,   // (B,N,N)
                      const float* __restrict__ cheb,  // (K,N,N)
                      _Float16*    __restrict__ rhs)   // ws, layout above
{
    const int kb = blockIdx.z;          // k*B + b
    const int k  = kb / CB;
    const int b  = kb % CB;
    const int n0 = blockIdx.y * 128;
    const int c0 = blockIdx.x * 128;

    // LDS transposed for fragment gathers: [row][m_local], stride 40 (80B, 16B-aligned)
    __shared__ _Float16 Wl[128][32 + 8];   // A rows (n) x K (m)
    __shared__ _Float16 Xl[128][32 + 8];   // B cols (c) x K (m)

    const int tid  = threadIdx.x;
    const int lane = tid & 31;
    const int wave = tid >> 5;          // 0..7
    const int wr   = wave >> 1;         // 0..3 : 32-row block
    const int wc   = wave & 1;          // 0..1 : 64-col block
    const int hl   = lane & 15;         // position within half-wave
    const int hs   = lane >> 4;         // half select

    v8f acc[2][4] = {};

    const float* __restrict__ chebK = cheb + (size_t)k * CN * CN;
    const float* __restrict__ attB  = att  + (size_t)b * CN * CN;
    const float* __restrict__ gsB   = gs   + (size_t)b * CN * CFT;

    // staging map: each thread owns row (tid&127) and two 8-wide m-segments
    const int srow = tid & 127;         // W: n offset / X: c offset
    const int sseg = tid >> 7;          // base segment (0..1); +2 for second

    // raw f32 staging registers (converted only at commit time)
    float creg[2][8], areg[2][8], xreg[2][8];

    // prologue: pure-load fetch of chunk 0
    #pragma unroll
    for (int s = 0; s < 2; ++s) {
        const int mb = (sseg + 2 * s) * 8;
        #pragma unroll
        for (int j = 0; j < 8; ++j) {
            const int m = mb + j;
            const size_t g = (size_t)m * CN + (n0 + srow);
            creg[s][j] = chebK[g];
            areg[s][j] = attB[g];
            xreg[s][j] = gsB[(size_t)m * CFT + (c0 + srow)];
        }
    }

    for (int m0 = 0; m0 < CN; m0 += 32) {
        // ---- commit: fused cheb*att multiply + f16 convert + 16B LDS stores
        #pragma unroll
        for (int s = 0; s < 2; ++s) {
            const int mb = (sseg + 2 * s) * 8;
            v8h w, x;
            #pragma unroll
            for (int j = 0; j < 8; ++j) {
                w[j] = (_Float16)(creg[s][j] * areg[s][j]);
                x[j] = (_Float16)xreg[s][j];
            }
            *(v8h*)&Wl[srow][mb] = w;
            *(v8h*)&Xl[srow][mb] = x;
        }
        __syncthreads();

        // ---- pure-load fetch of next chunk (no consuming VALU here, so the
        //      loadcnt wait lands at the NEXT commit, past the WMMA burst)
        if (m0 + 32 < CN) {
            #pragma unroll
            for (int s = 0; s < 2; ++s) {
                const int mb = (sseg + 2 * s) * 8;
                #pragma unroll
                for (int j = 0; j < 8; ++j) {
                    const int m = m0 + 32 + mb + j;
                    const size_t g = (size_t)m * CN + (n0 + srow);
                    creg[s][j] = chebK[g];
                    areg[s][j] = attB[g];
                    xreg[s][j] = gsB[(size_t)m * CFT + (c0 + srow)];
                }
            }
        }

        // ---- gather fragments per ISA 16-bit layouts (2x ds_load_b128 each)
        v16h afr[2], bfr[4];
        #pragma unroll
        for (int ai = 0; ai < 2; ++ai) {
            const int row = wr * 32 + ai * 16 + hl;
            const int kb8 = hs * 8;             // A: halves interleave K by 8
            #pragma unroll
            for (int e = 0; e < 16; ++e)
                afr[ai][e] = Wl[row][kb8 + e + (e >= 8 ? 8 : 0)];
        }
        #pragma unroll
        for (int bj = 0; bj < 4; ++bj) {
            const int col  = wc * 64 + bj * 16 + hl;
            const int kb16 = hs * 8;            // B: halves interleave K by 8
            #pragma unroll
            for (int e = 0; e < 16; ++e)
                bfr[bj][e] = Xl[col][kb16 + e + (e >= 8 ? 8 : 0)];
        }

        #pragma unroll
        for (int ai = 0; ai < 2; ++ai)
            #pragma unroll
            for (int bj = 0; bj < 4; ++bj)
                acc[ai][bj] = __builtin_amdgcn_wmma_f32_16x16x32_f16(
                    false, afr[ai], false, bfr[bj],
                    (short)0, acc[ai][bj], false, false);
        __syncthreads();
    }

    // ---- store rhs (f16) into [b][(n,t)][(k,f)] layout
    // D layout: col = hl (c index), row = r + hs*8 within tile
    _Float16* __restrict__ outB = rhs + (size_t)b * CNT * CKF;
    #pragma unroll
    for (int ai = 0; ai < 2; ++ai)
        #pragma unroll
        for (int bj = 0; bj < 4; ++bj) {
            const int cgl = c0 + wc * 64 + bj * 16 + hl;   // c = f*T + t
            const int f = cgl / CT, t = cgl % CT;
            #pragma unroll
            for (int r = 0; r < 8; ++r) {
                const int ngl = n0 + wr * 32 + ai * 16 + r + hs * 8;
                outB[(size_t)(ngl * CT + t) * CKF + (k * CF + f)] =
                    (_Float16)acc[ai][bj][r];
            }
        }
}

// ---------------------------------------------------------------------------
// Stage 2: out[b,n,o,t] = relu( sum_{kf} rhs[b,(n,t),kf] * theta[kf,o] )
// GEMM per b: rows = (n,t) (12288), K = 192, cols = O (64).
// WG: 8 waves x 32 rows = 256 rows; wave tile 32x64 -> 2x4 accumulators.
// A fragments: two contiguous 16B global loads per lane from f16 workspace.
// ---------------------------------------------------------------------------
__global__ __launch_bounds__(256)
void cheb_stage2_wmma(const _Float16* __restrict__ rhs,   // [b][nt][kf] f16
                      const float*    __restrict__ theta, // (K,F,O)
                      float*          __restrict__ out)   // (B,N,O,T)
{
    const int b    = blockIdx.y;
    const int row0 = blockIdx.x * 256;       // over n*T rows

    __shared__ _Float16 Tl[64][32 + 8];      // [o][kk_local], stride 40

    const int tid  = threadIdx.x;
    const int lane = tid & 31;
    const int wave = tid >> 5;
    const int hl   = lane & 15;
    const int hs   = lane >> 4;

    const _Float16* __restrict__ rhsB = rhs + (size_t)b * CNT * CKF;

    v8f acc[2][4] = {};

    for (int kc = 0; kc < CKF; kc += 32) {
        // ---- stage theta chunk: Tl[oo][mm] = theta[(kc+mm)*O + oo], 16B stores
        {
            const int oo  = tid & 63;
            const int seg = tid >> 6;           // 0..3
            v8h th;
            #pragma unroll
            for (int j = 0; j < 8; ++j)
                th[j] = (_Float16)theta[(size_t)(kc + seg * 8 + j) * CO + oo];
            *(v8h*)&Tl[oo][seg * 8] = th;
        }
        __syncthreads();

        // ---- A fragments: 2x contiguous 16B loads per lane per fragment
        v16h afr[2];
        #pragma unroll
        for (int ai = 0; ai < 2; ++ai) {
            const int rgl = row0 + wave * 32 + ai * 16 + hl;
            const v8h* p = (const v8h*)(rhsB + (size_t)rgl * CKF + kc + hs * 8);
            const v8h lo = p[0];    // K = kc + hs*8 + [0..7]
            const v8h hi = p[2];    // K = kc + hs*8 + 16 + [0..7]
            afr[ai] = __builtin_shufflevector(lo, hi,
                0, 1, 2, 3, 4, 5, 6, 7, 8, 9, 10, 11, 12, 13, 14, 15);
        }

        #pragma unroll
        for (int bj = 0; bj < 4; ++bj) {
            v16h bfr;
            const int col = bj * 16 + hl;
            #pragma unroll
            for (int e = 0; e < 16; ++e)
                bfr[e] = Tl[col][hs * 8 + e + (e >= 8 ? 8 : 0)];
            #pragma unroll
            for (int ai = 0; ai < 2; ++ai)
                acc[ai][bj] = __builtin_amdgcn_wmma_f32_16x16x32_f16(
                    false, afr[ai], false, bfr, (short)0, acc[ai][bj], false, false);
        }
        __syncthreads();
    }

    // ---- fused ReLU + store f32 to (B,N,O,T)
    #pragma unroll
    for (int ai = 0; ai < 2; ++ai)
        #pragma unroll
        for (int bj = 0; bj < 4; ++bj) {
            const int o = bj * 16 + hl;
            #pragma unroll
            for (int r = 0; r < 8; ++r) {
                const int rgl = row0 + wave * 32 + ai * 16 + r + hs * 8;
                const int n = rgl / CT, t = rgl % CT;
                out[(((size_t)b * CN + n) * CO + o) * CT + t] =
                    fmaxf(acc[ai][bj][r], 0.0f);
            }
        }
}

extern "C" void kernel_launch(void* const* d_in, const int* in_sizes, int n_in,
                              void* d_out, int out_size, void* d_ws, size_t ws_size,
                              hipStream_t stream) {
    const float* gs    = (const float*)d_in[0];  // graph_signal (B,N,F,T)
    const float* att   = (const float*)d_in[1];  // spatial_attention (B,N,N)
    const float* cheb  = (const float*)d_in[2];  // cheb (K,N,N)
    const float* theta = (const float*)d_in[3];  // theta (K,F,O)
    float*       outp  = (float*)d_out;          // (B,N,O,T) f32
    _Float16*    rhs   = (_Float16*)d_ws;        // [b][nt][kf] f16 = 37.75 MB

    // Stage 1: grid = (c-tiles, n-tiles, K*B)
    dim3 g1(CFT / 128, CN / 128, CK * CB);
    cheb_stage1_wmma<<<g1, 256, 0, stream>>>(gs, att, cheb, rhs);

    // Stage 2: grid = (row-tiles over n*T, B)
    dim3 g2(CNT / 256, CB, 1);
    cheb_stage2_wmma<<<g2, 256, 0, stream>>>(rhs, theta, outp);
}
